// CircleLoss_80547816669830
// MI455X (gfx1250) — matrix operand order repaired and checked by reading, not compiled, not run on hardware
//
#include <hip/hip_runtime.h>

typedef __attribute__((ext_vector_type(16))) _Float16 v16h;
typedef __attribute__((ext_vector_type(8)))  _Float16 v8h;
typedef __attribute__((ext_vector_type(8)))  float    v8f;

#define N_ROWS 8192
#define DIM    512
#define SCALE_F  64.0f
#define MARGIN_F 0.35f

#define BLK   128                       // block macro-tile: 128x128
#define TB    (N_ROWS / BLK)            // 64 block-tiles per dimension
#define NBLK  (TB * (TB + 1) / 2)       // 2080 upper-triangular blocks

// ---------------------------------------------------------------------------
// Kernel 1: row-wise L2 normalize (clamped at 1e-8) f32 -> f16 workspace
// ---------------------------------------------------------------------------
__global__ void __launch_bounds__(256) normalize_rows(const float* __restrict__ emb,
                                                      _Float16* __restrict__ E) {
    const int row = blockIdx.x;
    const int t   = threadIdx.x;
    const float* x = emb + (size_t)row * DIM;
    float a0 = x[t];
    float a1 = x[t + 256];

    __shared__ float red[256];
    red[t] = a0 * a0 + a1 * a1;
    __syncthreads();
    for (int s = 128; s > 0; s >>= 1) {
        if (t < s) red[t] += red[t + s];
        __syncthreads();
    }
    const float inv = 1.0f / fmaxf(sqrtf(red[0]), 1e-8f);

    _Float16* o = E + (size_t)row * DIM;
    o[t]       = (_Float16)(a0 * inv);
    o[t + 256] = (_Float16)(a1 * inv);
}

// ---------------------------------------------------------------------------
// Kernel 2: upper-block-triangular fused Gram -> circle loss -> block partial.
//
// Block = 128 threads = 4 wave32 arranged 2(i) x 2(j); each wave owns a
// 64x64 macro-tile = 4 i-tiles x 4 j-tiles of 16x16 -> 16 f32 accumulators.
// Per K=32 step: 8 A loads + 8 B loads feed 16 v_wmma_f32_16x16x32_f16
// (1.0 b128 loads per WMMA, ~32 FLOP/byte from L0).
//
// Fragment layouts (ISA 7.12.2 / 7.12.4, 16-bit):
//   A 16x32: lane L (M=L%16) holds K = a0..a0+7 and a0+16..a0+23, a0 = 8*(L/16)
//   B 32x16: lane L (N=L%16) holds K = b0..b0+7 and b0+16..b0+23, b0 = 8*(L/16)
// For E*E^T, B column n is E row (j0+n): both fragments are b128 row loads.
// ---------------------------------------------------------------------------
__global__ void __launch_bounds__(128) gram_loss(const _Float16* __restrict__ E,
                                                 const int*      __restrict__ labels,
                                                 float*          __restrict__ partials) {
    // ---- decode upper-triangular block index: t = bi*(bi+1)/2 + bj, bj<=bi
    const int t = blockIdx.x;
    int bi = (int)((sqrtf(8.0f * (float)t + 1.0f) - 1.0f) * 0.5f);
    while ((bi + 1) * (bi + 2) / 2 <= t) ++bi;
    while (bi * (bi + 1) / 2 > t) --bi;
    const int bj = t - bi * (bi + 1) / 2;
    const int rb = bj * BLK;            // i-rows of this block (rb <= cb)
    const int cb = bi * BLK;            // j-cols of this block
    const float weight = (bi == bj) ? 1.0f : 2.0f;

    const int lane = threadIdx.x & 31;
    const int wave = threadIdx.x >> 5;  // 0..3
    const int wi   = wave >> 1;         // 0..1 -> 64-row slab
    const int wj   = wave & 1;          // 0..1 -> 64-col slab
    const int sel  = lane >> 4;
    const int lm   = lane & 15;

    const int i_base = rb + wi * 64;    // wave's first i row
    const int j_base = cb + wj * 64;    // wave's first j col

    const _Float16* rowA[4];
    const _Float16* rowB[4];
#pragma unroll
    for (int it = 0; it < 4; ++it) rowA[it] = E + (size_t)(i_base + it * 16 + lm) * DIM;
#pragma unroll
    for (int jt = 0; jt < 4; ++jt) rowB[jt] = E + (size_t)(j_base + jt * 16 + lm) * DIM;
    const int koffA = sel * 8;          // A: runs at koffA and koffA+16
    const int koffB = sel * 8;          // B: runs at koffB and koffB+16

    v8f acc[4][4] = {};
#pragma unroll 2
    for (int kk = 0; kk < DIM; kk += 32) {
        v16h a[4], b[4];
#pragma unroll
        for (int it = 0; it < 4; ++it) {
            v8h lo = *(const v8h*)(rowA[it] + kk + koffA);
            v8h hi = *(const v8h*)(rowA[it] + kk + 16 + koffA);
            a[it] = __builtin_shufflevector(lo, hi, 0,1,2,3,4,5,6,7,8,9,10,11,12,13,14,15);
        }
#pragma unroll
        for (int jt = 0; jt < 4; ++jt) {
            v8h lo = *(const v8h*)(rowB[jt] + kk + koffB);
            v8h hi = *(const v8h*)(rowB[jt] + kk + 16 + koffB);
            b[jt] = __builtin_shufflevector(lo, hi, 0,1,2,3,4,5,6,7,8,9,10,11,12,13,14,15);
        }
#pragma unroll
        for (int it = 0; it < 4; ++it)
#pragma unroll
            for (int jt = 0; jt < 4; ++jt)
                acc[it][jt] = __builtin_amdgcn_wmma_f32_16x16x32_f16(
                    false, a[it], false, b[jt],
                    (short)0, acc[it][jt], false, false);
    }

    // ---- epilogue: acc[it][jt][r] = sim(i_base + it*16 + 8*sel + r, j_base + jt*16 + lm)
    int labj[4], labi[4][8];
#pragma unroll
    for (int jt = 0; jt < 4; ++jt) labj[jt] = labels[j_base + jt * 16 + lm];
#pragma unroll
    for (int it = 0; it < 4; ++it)
#pragma unroll
        for (int r = 0; r < 8; ++r) labi[it][r] = labels[i_base + it * 16 + sel * 8 + r];

    float lsum = 0.0f;
#pragma unroll
    for (int it = 0; it < 4; ++it)
#pragma unroll
        for (int jt = 0; jt < 4; ++jt)
#pragma unroll
            for (int r = 0; r < 8; ++r) {
                const float sim = acc[it][jt][r];
                const float sgn = (labi[it][r] == labj[jt]) ? sim : -sim;
                // log1p(exp(-SCALE*(sgn - MARGIN))) = softplus(x), stable form
                const float x = SCALE_F * (MARGIN_F - sgn);
                lsum += fmaxf(x, 0.0f) + log1pf(__expf(-fabsf(x)));
            }
    lsum *= weight;

    __shared__ float red[128];
    red[threadIdx.x] = lsum;
    __syncthreads();
    for (int s = 64; s > 0; s >>= 1) {
        if (threadIdx.x < s) red[threadIdx.x] += red[threadIdx.x + s];
        __syncthreads();
    }
    if (threadIdx.x == 0) partials[blockIdx.x] = red[0];
}

// ---------------------------------------------------------------------------
// Kernel 3: deterministic final reduction of block partials -> mean
// ---------------------------------------------------------------------------
__global__ void __launch_bounds__(256) final_reduce(const float* __restrict__ partials,
                                                    float* __restrict__ out) {
    __shared__ float red[256];
    float s = 0.0f;
    for (int idx = threadIdx.x; idx < NBLK; idx += 256) s += partials[idx];
    red[threadIdx.x] = s;
    __syncthreads();
    for (int st = 128; st > 0; st >>= 1) {
        if (threadIdx.x < st) red[threadIdx.x] += red[threadIdx.x + st];
        __syncthreads();
    }
    if (threadIdx.x == 0)
        out[0] = red[0] * (1.0f / ((float)N_ROWS * (float)N_ROWS));
}

extern "C" void kernel_launch(void* const* d_in, const int* in_sizes, int n_in,
                              void* d_out, int out_size, void* d_ws, size_t ws_size,
                              hipStream_t stream) {
    const float* emb    = (const float*)d_in[0];
    const int*   labels = (const int*)d_in[1];
    float*       out    = (float*)d_out;

    _Float16* E        = (_Float16*)d_ws;                                   // 8 MB f16 matrix
    float*    partials = (float*)((char*)d_ws + (size_t)N_ROWS * DIM * 2);  // block partials

    normalize_rows<<<N_ROWS, 256, 0, stream>>>(emb, E);
    gram_loss<<<NBLK, 128, 0, stream>>>(E, labels, partials);
    final_reduce<<<1, 256, 0, stream>>>(partials, out);
}
